// VisionTransformer_84576495993014
// MI455X (gfx1250) — matrix operand order repaired and checked by reading, not compiled, not run on hardware
//
#include <hip/hip_runtime.h>
#include <cmath>

typedef unsigned short u16;
typedef __attribute__((ext_vector_type(16))) __bf16 v16bf;
typedef __attribute__((ext_vector_type(8)))  float   v8f;

// ---------------- bf16 helpers (bit-level) ----------------
__device__ inline u16 f2bf(float f) {
  union { float f; unsigned int u; } v; v.f = f;
  unsigned int u = v.u;
  unsigned int r = (u + 0x7FFFu + ((u >> 16) & 1u)) >> 16;
  return (u16)r;
}
__device__ inline u16 toBits(float v) { return f2bf(v); }
__device__ inline u16 toBits(u16 v)   { return v; }
__device__ inline void storeC(float* p, float v) { *p = v; }
__device__ inline void storeC(u16* p,   float v) { *p = f2bf(v); }

// ======================= WMMA GEMM =======================
// C[M,N] = A[M,K] * B'[N,K]  (both bf16, B pre-transposed), f32 accum.
// Requirements: K % 32 == 0; lda/ldb multiples of 8; row OOB handled by guards.
// 128x128 block tile, K-step 32, double-buffered LDS, 256 threads = 8 waves,
// each wave owns a 64x32 patch = 4x2 v_wmma_f32_16x16x32_bf16 accumulators.
#define BM 128
#define BN 128
#define BK 32
#define LDSS 40

template <typename TC, bool RELU, bool RESADD>
__global__ __launch_bounds__(256)
void gemm_wmma_kernel(const u16* __restrict__ A,
                      const u16* __restrict__ B,     // [N][K] bf16
                      const float* __restrict__ bias,
                      const float* __restrict__ res,
                      TC* __restrict__ C,
                      int M, int N, int K,
                      int lda, int ldb, int ldc,
                      long long sA, long long sB, long long sC,
                      float scale)
{
  __shared__ u16 As[2][BM][LDSS];   // [m][k]
  __shared__ u16 Bs[2][BN][LDSS];   // [n][k]

  const int tid  = threadIdx.x;
  const int lane = tid & 31;
  const int wave = tid >> 5;
  const int wm   = wave >> 2;   // 0..1 -> 64 rows
  const int wn   = wave & 3;    // 0..3 -> 32 cols
  const int lrow = lane & 15;
  const int lhi  = lane >> 4;

  const int row0 = blockIdx.y * BM;
  const int col0 = blockIdx.x * BN;
  const long long z = blockIdx.z;
  A += z * sA;  B += z * sB;  C += z * sC;
  if (RESADD) res += z * sC;

  // staging map: thread t covers rows (t>>2) and (t>>2)+64, 8 elems at (t&3)*8
  const int rS   = tid >> 2;
  const int offS = (tid & 3) * 8;

  v8f zero = {};
  v8f acc[4][2];
#pragma unroll
  for (int i = 0; i < 4; ++i)
#pragma unroll
    for (int j = 0; j < 2; ++j) acc[i][j] = zero;

  const int ksteps = K / BK;

  uint4 aReg[2], bReg[2];
  // ---- prologue: fetch tile 0 ----
  {
#pragma unroll
    for (int c = 0; c < 2; ++c) {
      int gr = row0 + rS + 64 * c;
      aReg[c] = make_uint4(0u, 0u, 0u, 0u);
      if (gr < M) aReg[c] = *reinterpret_cast<const uint4*>(A + (long long)gr * lda + offS);
      int gn = col0 + rS + 64 * c;
      bReg[c] = make_uint4(0u, 0u, 0u, 0u);
      if (gn < N) bReg[c] = *reinterpret_cast<const uint4*>(B + (long long)gn * ldb + offS);
    }
#pragma unroll
    for (int c = 0; c < 2; ++c) {
      *reinterpret_cast<uint4*>(&As[0][rS + 64 * c][offS]) = aReg[c];
      *reinterpret_cast<uint4*>(&Bs[0][rS + 64 * c][offS]) = bReg[c];
    }
  }
  __syncthreads();

  for (int kt = 0; kt < ksteps; ++kt) {
    const int cur = kt & 1;
    const int nxt = cur ^ 1;

    // ---- issue next tile's global loads early (overlap with WMMA) ----
    if (kt + 1 < ksteps) {
      const int k0 = (kt + 1) * BK;
#pragma unroll
      for (int c = 0; c < 2; ++c) {
        int gr = row0 + rS + 64 * c;
        aReg[c] = make_uint4(0u, 0u, 0u, 0u);
        if (gr < M) aReg[c] = *reinterpret_cast<const uint4*>(A + (long long)gr * lda + k0 + offS);
        int gn = col0 + rS + 64 * c;
        bReg[c] = make_uint4(0u, 0u, 0u, 0u);
        if (gn < N) bReg[c] = *reinterpret_cast<const uint4*>(B + (long long)gn * ldb + k0 + offS);
      }
      if (kt + 2 < ksteps) {   // speculative prefetch of tile kt+2 -> global_prefetch_b8
        __builtin_prefetch(A + (long long)(row0 + rS) * lda + (kt + 2) * BK + offS, 0, 0);
        __builtin_prefetch(B + (long long)(col0 + rS) * ldb + (kt + 2) * BK + offS, 0, 0);
      }
    }

    // ---- fragment loads from LDS (b128, ISA 16-bit A/B lane layouts) ----
    union Frag { uint4 q[2]; v16bf v; };
    Frag af[4], bf[2];
#pragma unroll
    for (int mt = 0; mt < 4; ++mt) {
      const u16* ap = &As[cur][wm * 64 + mt * 16 + lrow][lhi * 8];
      af[mt].q[0] = *reinterpret_cast<const uint4*>(ap);        // K 0..7   (8..15 hi lanes)
      af[mt].q[1] = *reinterpret_cast<const uint4*>(ap + 16);   // K 16..23 (24..31 hi lanes)
    }
#pragma unroll
    for (int nt = 0; nt < 2; ++nt) {
      const u16* bp = &Bs[cur][wn * 32 + nt * 16 + lrow][lhi * 8];
      bf[nt].q[0] = *reinterpret_cast<const uint4*>(bp);
      bf[nt].q[1] = *reinterpret_cast<const uint4*>(bp + 16);
    }

    // ---- 8 WMMA per wave per K-step ----
#pragma unroll
    for (int mt = 0; mt < 4; ++mt)
#pragma unroll
      for (int nt = 0; nt < 2; ++nt)
        acc[mt][nt] = __builtin_amdgcn_wmma_f32_16x16x32_bf16(
            false, af[mt].v, false, bf[nt].v, (short)0, acc[mt][nt], false, false);

    // ---- commit next tile into the alternate LDS buffer ----
    if (kt + 1 < ksteps) {
#pragma unroll
      for (int c = 0; c < 2; ++c) {
        *reinterpret_cast<uint4*>(&As[nxt][rS + 64 * c][offS]) = aReg[c];
        *reinterpret_cast<uint4*>(&Bs[nxt][rS + 64 * c][offS]) = bReg[c];
      }
    }
    __syncthreads();
  }

  // ---- epilogue: acc VGPR r -> row base+lhi*8+r, col base+lrow ----
#pragma unroll
  for (int mt = 0; mt < 4; ++mt) {
#pragma unroll
    for (int nt = 0; nt < 2; ++nt) {
      int col = col0 + wn * 32 + nt * 16 + lrow;
      if (col >= N) continue;
      float bv = bias ? bias[col] : 0.0f;
      v8f a8 = acc[mt][nt];
#pragma unroll
      for (int r = 0; r < 8; ++r) {
        int row = row0 + wm * 64 + mt * 16 + lhi * 8 + r;
        if (row >= M) continue;
        float v = a8[r] * scale + bv;
        if (RELU)   v = v > 0.0f ? v : 0.0f;
        if (RESADD) v += res[(long long)row * ldc + col];
        storeC(&C[(long long)row * ldc + col], v);
      }
    }
  }
}

// ============ tiled transpose+convert: out[c][r] = in[r][c] (bf16), zero-pad r>=rows ======
template <typename TI>
__global__ __launch_bounds__(256)
void transposeT_kernel(const TI* __restrict__ in, u16* __restrict__ out,
                       int rows, int cols, int rowsPad,
                       long long inStride, long long outStride)
{
  __shared__ u16 tile[32][33];
  in  += (long long)blockIdx.z * inStride;
  out += (long long)blockIdx.z * outStride;
  const int c0 = blockIdx.x * 32;
  const int r0 = blockIdx.y * 32;
  const int tx = threadIdx.x & 31;
  const int ty = threadIdx.x >> 5;       // 0..7
#pragma unroll
  for (int i = 0; i < 4; ++i) {
    int rl = ty * 4 + i;
    int r = r0 + rl, c = c0 + tx;
    u16 v = 0;
    if (r < rows && c < cols) v = toBits(in[(long long)r * cols + c]);
    tile[rl][tx] = v;
  }
  __syncthreads();
#pragma unroll
  for (int i = 0; i < 4; ++i) {
    int cl = ty * 4 + i;
    int c = c0 + cl, r = r0 + tx;
    if (c < cols && r < rowsPad)
      out[(long long)c * rowsPad + r] = tile[tx][cl];
  }
}

// ---------------- elementwise fp32 -> bf16 ----------------
__global__ __launch_bounds__(256)
void convert_bf16_kernel(const float* __restrict__ in, u16* __restrict__ out, int total)
{
  int t = blockIdx.x * 256 + threadIdx.x;
  if (t < total) out[t] = f2bf(in[t]);
}

// ---------------- im2col: x[B,3,224,224] -> A0 bf16 [B*196, 768] -----
__global__ __launch_bounds__(256)
void im2col_kernel(const float* __restrict__ x, u16* __restrict__ a0, int total)
{
  int t = blockIdx.x * 256 + threadIdx.x;
  if (t >= total) return;
  int row = t / 768, col = t % 768;
  int b = row / 196, p = row % 196;
  int py = p / 14, px = p % 14;
  int c = col / 256, rem = col % 256;
  int ky = rem / 16, kx = rem % 16;
  float v = x[(((long long)b * 3 + c) * 224 + (py * 16 + ky)) * 224 + (px * 16 + kx)];
  a0[t] = f2bf(v);
}

// ---------------- assemble: h[b,s,:] = (s==0?cls:pe[b,s-1]) + pos[s] ------------------
__global__ __launch_bounds__(256)
void assemble_kernel(const float* __restrict__ pe, const float* __restrict__ cls,
                     const float* __restrict__ pos, float* __restrict__ h, int total)
{
  int t = blockIdx.x * 256 + threadIdx.x;
  if (t >= total) return;
  int b = t / (197 * 768);
  int r = t % (197 * 768);
  int s = r / 768, e = r % 768;
  float v = (s == 0) ? cls[e] : pe[((long long)b * 196 + (s - 1)) * 768 + e];
  h[t] = v + pos[(long long)s * 768 + e];
}

// ---------------- LayerNorm (E=768): fp32 in -> bf16 out, one block per row ----------
__global__ __launch_bounds__(256)
void ln_kernel(const float* __restrict__ x, const float* __restrict__ s,
               const float* __restrict__ b, u16* __restrict__ out,
               long long strideIn, long long strideOut)
{
  __shared__ float red[256];
  const int tid = threadIdx.x;
  const float* xr = x + (long long)blockIdx.x * strideIn;
  float v0 = xr[tid], v1 = xr[tid + 256], v2 = xr[tid + 512];
  red[tid] = v0 + v1 + v2;
  __syncthreads();
  for (int o = 128; o > 0; o >>= 1) { if (tid < o) red[tid] += red[tid + o]; __syncthreads(); }
  float mean = red[0] * (1.0f / 768.0f);
  __syncthreads();
  float d0 = v0 - mean, d1 = v1 - mean, d2 = v2 - mean;
  red[tid] = d0 * d0 + d1 * d1 + d2 * d2;
  __syncthreads();
  for (int o = 128; o > 0; o >>= 1) { if (tid < o) red[tid] += red[tid + o]; __syncthreads(); }
  float rstd = rsqrtf(red[0] * (1.0f / 768.0f) + 1e-5f);
  u16* orow = out + (long long)blockIdx.x * strideOut;
  orow[tid]       = f2bf(d0 * rstd * s[tid]       + b[tid]);
  orow[tid + 256] = f2bf(d1 * rstd * s[tid + 256] + b[tid + 256]);
  orow[tid + 512] = f2bf(d2 * rstd * s[tid + 512] + b[tid + 512]);
}

// ---------------- softmax over 197 cols, ld 224, zero-pad 197..223 -------------------
__global__ __launch_bounds__(256)
void softmax_kernel(const float* __restrict__ att, u16* __restrict__ p)
{
  __shared__ float red[256];
  const int c = threadIdx.x;
  const float* ar = att + (long long)blockIdx.x * 224;
  float v = (c < 197) ? ar[c] : -3.0e38f;
  red[c] = v;
  __syncthreads();
  for (int o = 128; o > 0; o >>= 1) { if (c < o) red[c] = fmaxf(red[c], red[c + o]); __syncthreads(); }
  float mx = red[0];
  __syncthreads();
  float e = (c < 197) ? __expf(v - mx) : 0.0f;
  red[c] = e;
  __syncthreads();
  for (int o = 128; o > 0; o >>= 1) { if (c < o) red[c] += red[c + o]; __syncthreads(); }
  float inv = 1.0f / red[0];
  if (c < 224)
    p[(long long)blockIdx.x * 224 + c] = (c < 197) ? f2bf(e * inv) : (u16)0;
}

// =================================== host launcher ===================================
extern "C" void kernel_launch(void* const* d_in, const int* in_sizes, int n_in,
                              void* d_out, int out_size, void* d_ws, size_t ws_size,
                              hipStream_t stream)
{
  (void)in_sizes; (void)n_in; (void)out_size; (void)ws_size;
  const int E = 768, EH = 3072, L = 12, S = 197, SP = 224, NP = 196, Bsz = 32, OUTW = 512;
  const int ROWS = Bsz * S;    // 6304
  const int PROWS = Bsz * NP;  // 6272

  const float* x      = (const float*)d_in[0];
  const float* conv_w = (const float*)d_in[1];
  const float* cls    = (const float*)d_in[2];
  const float* pos    = (const float*)d_in[3];
  const float* ln1_s  = (const float*)d_in[4];
  const float* ln1_b  = (const float*)d_in[5];
  const float* Wq     = (const float*)d_in[6];
  const float* bq     = (const float*)d_in[7];
  const float* Wk     = (const float*)d_in[8];
  const float* bk     = (const float*)d_in[9];
  const float* Wv     = (const float*)d_in[10];
  const float* bv     = (const float*)d_in[11];
  const float* Wo     = (const float*)d_in[12];
  const float* bo     = (const float*)d_in[13];
  const float* ln2_s  = (const float*)d_in[14];
  const float* ln2_b  = (const float*)d_in[15];
  const float* W1     = (const float*)d_in[16];
  const float* b1     = (const float*)d_in[17];
  const float* W2     = (const float*)d_in[18];
  const float* b2     = (const float*)d_in[19];
  const float* lnf_s  = (const float*)d_in[20];
  const float* lnf_b  = (const float*)d_in[21];
  const float* projW  = (const float*)d_in[22];

  // ---- workspace carve (all chunks are 256B-multiples) ----
  char* base = (char*)d_ws;
  size_t off = 0;
  u16*   a0    = (u16*)(base + off);   off += (size_t)PROWS * E * 2;     // im2col bf16
  float* pe    = (float*)(base + off); off += (size_t)PROWS * E * 4;     // patch-embed f32
  float* h     = (float*)(base + off); off += (size_t)ROWS * E * 4;      // residual stream
  u16*   nbuf  = (u16*)(base + off);   off += (size_t)ROWS * E * 2;      // LN out bf16
  u16*   qb    = (u16*)(base + off);   off += (size_t)ROWS * EH * 2;
  u16*   kb    = (u16*)(base + off);   off += (size_t)ROWS * EH * 2;
  u16*   vb    = (u16*)(base + off);   off += (size_t)ROWS * EH * 2;
  u16*   vbT   = (u16*)(base + off);   off += (size_t)Bsz * EH * SP * 2; // V^T padded bf16
  float* attf  = (float*)(base + off); off += (size_t)ROWS * SP * 4;     // scores f32
  u16*   prob  = (u16*)(base + off);   off += (size_t)ROWS * SP * 2;     // softmax bf16 (padded)
  u16*   zb    = (u16*)(base + off);   off += (size_t)ROWS * EH * 2;     // att@v bf16
  u16*   mb    = (u16*)(base + off);   off += (size_t)ROWS * EH * 2;     // MLP hidden bf16
  u16*   clsn  = (u16*)(base + off);   off += (size_t)Bsz * E * 2;
  u16*   convB = (u16*)(base + off);   off += (size_t)E * E * 2;         // conv_w bf16 [N][K]
  u16*   projT = (u16*)(base + off);   off += (size_t)OUTW * E * 2;      // proj^T bf16
  u16*   wqT   = (u16*)(base + off);   off += (size_t)EH * E * 2;        // [3072][768]
  u16*   wkT   = (u16*)(base + off);   off += (size_t)EH * E * 2;
  u16*   wvT   = (u16*)(base + off);   off += (size_t)EH * E * 2;
  u16*   woT   = (u16*)(base + off);   off += (size_t)E * EH * 2;        // [768][3072]
  u16*   w1T   = (u16*)(base + off);   off += (size_t)EH * E * 2;
  u16*   w2T   = (u16*)(base + off);   off += (size_t)E * EH * 2;

  const float inv_scale = 1.0f / powf((float)E, 0.25f);
  const dim3 blk(256);

  // 1) im2col + weight prep (conv_w already [N][K]; proj needs transpose)
  {
    int total = PROWS * E;
    im2col_kernel<<<dim3((total + 255) / 256), blk, 0, stream>>>(x, a0, total);
    convert_bf16_kernel<<<dim3((E * E + 255) / 256), blk, 0, stream>>>(conv_w, convB, E * E);
    transposeT_kernel<float><<<dim3(OUTW / 32, E / 32, 1), blk, 0, stream>>>(
        projW, projT, E, OUTW, E, 0, 0);
  }
  // 2) patch embed GEMM: pe = a0 @ conv_w^T
  gemm_wmma_kernel<float, false, false>
      <<<dim3(E / 128, (PROWS + 127) / 128, 1), blk, 0, stream>>>(
      a0, convB, nullptr, nullptr, pe,
      PROWS, E, E, E, E, E, 0, 0, 0, 1.0f);
  // 3) assemble h = concat(cls, pe) + pos
  {
    int total = ROWS * E;
    assemble_kernel<<<dim3((total + 255) / 256), blk, 0, stream>>>(pe, cls, pos, h, total);
  }

  for (int l = 0; l < L; ++l) {
    const float* l1s = ln1_s + (size_t)l * E;
    const float* l1b = ln1_b + (size_t)l * E;
    const float* wq  = Wq + (size_t)l * E * EH;  const float* bq_ = bq + (size_t)l * EH;
    const float* wk  = Wk + (size_t)l * E * EH;  const float* bk_ = bk + (size_t)l * EH;
    const float* wv  = Wv + (size_t)l * E * EH;  const float* bv_ = bv + (size_t)l * EH;
    const float* wo  = Wo + (size_t)l * EH * E;  const float* bo_ = bo + (size_t)l * E;
    const float* l2s = ln2_s + (size_t)l * E;
    const float* l2b = ln2_b + (size_t)l * E;
    const float* w1  = W1 + (size_t)l * E * EH;  const float* b1_ = b1 + (size_t)l * EH;
    const float* w2  = W2 + (size_t)l * EH * E;  const float* b2_ = b2 + (size_t)l * E;

    // weight prep: bf16 [N][K] transposes for this layer
    dim3 gT1(EH / 32, E / 32, 1);   // in [768][3072] -> out [3072][768]
    dim3 gT2(E / 32, EH / 32, 1);   // in [3072][768] -> out [768][3072]
    transposeT_kernel<float><<<gT1, blk, 0, stream>>>(wq, wqT, E, EH, E, 0, 0);
    transposeT_kernel<float><<<gT1, blk, 0, stream>>>(wk, wkT, E, EH, E, 0, 0);
    transposeT_kernel<float><<<gT1, blk, 0, stream>>>(wv, wvT, E, EH, E, 0, 0);
    transposeT_kernel<float><<<gT2, blk, 0, stream>>>(wo, woT, EH, E, EH, 0, 0);
    transposeT_kernel<float><<<gT1, blk, 0, stream>>>(w1, w1T, E, EH, E, 0, 0);
    transposeT_kernel<float><<<gT2, blk, 0, stream>>>(w2, w2T, EH, E, EH, 0, 0);

    // LN1
    ln_kernel<<<dim3(ROWS), blk, 0, stream>>>(h, l1s, l1b, nbuf, E, E);

    // Q, K, V : [6304,768] @ [768,3072] -> bf16
    dim3 gQKV(EH / 128, (ROWS + 127) / 128, 1);
    gemm_wmma_kernel<u16, false, false><<<gQKV, blk, 0, stream>>>(
        nbuf, wqT, bq_, nullptr, qb, ROWS, EH, E, E, E, EH, 0, 0, 0, 1.0f);
    gemm_wmma_kernel<u16, false, false><<<gQKV, blk, 0, stream>>>(
        nbuf, wkT, bk_, nullptr, kb, ROWS, EH, E, E, E, EH, 0, 0, 0, 1.0f);
    gemm_wmma_kernel<u16, false, false><<<gQKV, blk, 0, stream>>>(
        nbuf, wvT, bv_, nullptr, vb, ROWS, EH, E, E, E, EH, 0, 0, 0, 1.0f);

    // scores: att[b] = q[b] @ k[b]^T  (K matrix already [N=197][K=3072])
    gemm_wmma_kernel<float, false, false>
        <<<dim3(2, 2, Bsz), blk, 0, stream>>>(
        qb, kb, nullptr, nullptr, attf,
        S, S, EH, EH, EH, SP,
        (long long)S * EH, (long long)S * EH, (long long)S * SP, 1.0f);

    // softmax rows (padded to 224, zeros beyond 197)
    softmax_kernel<<<dim3(ROWS), blk, 0, stream>>>(attf, prob);

    // V^T per batch: [197][3072] -> [3072][224] (zero-padded K)
    transposeT_kernel<u16><<<dim3(EH / 32, SP / 32, Bsz), blk, 0, stream>>>(
        vb, vbT, S, EH, SP, (long long)S * EH, (long long)EH * SP);

    // z[b] = (prob[b] @ v[b]) * inv_scale -> bf16
    gemm_wmma_kernel<u16, false, false>
        <<<dim3(EH / 128, 2, Bsz), blk, 0, stream>>>(
        prob, vbT, nullptr, nullptr, zb,
        S, EH, SP, SP, SP, EH,
        (long long)S * SP, (long long)EH * SP, (long long)S * EH, inv_scale);

    // h += z @ Wo + bo
    gemm_wmma_kernel<float, false, true>
        <<<dim3(E / 128, (ROWS + 127) / 128, 1), blk, 0, stream>>>(
        zb, woT, bo_, h, h, ROWS, E, EH, EH, EH, E, 0, 0, 0, 1.0f);

    // LN2
    ln_kernel<<<dim3(ROWS), blk, 0, stream>>>(h, l2s, l2b, nbuf, E, E);

    // m = relu(n @ W1 + b1) -> bf16
    gemm_wmma_kernel<u16, true, false><<<gQKV, blk, 0, stream>>>(
        nbuf, w1T, b1_, nullptr, mb, ROWS, EH, E, E, E, EH, 0, 0, 0, 1.0f);

    // h += m @ W2 + b2
    gemm_wmma_kernel<float, false, true>
        <<<dim3(E / 128, (ROWS + 127) / 128, 1), blk, 0, stream>>>(
        mb, w2T, b2_, h, h, ROWS, E, EH, EH, EH, E, 0, 0, 0, 1.0f);
  }

  // final LN on cls rows, then projection -> d_out [32,512]
  ln_kernel<<<dim3(Bsz), blk, 0, stream>>>(h, lnf_s, lnf_b, clsn, (long long)S * E, E);

  gemm_wmma_kernel<float, false, false>
      <<<dim3(OUTW / 128, 1, 1), blk, 0, stream>>>(
      clsn, projT, nullptr, nullptr, (float*)d_out,
      Bsz, OUTW, E, E, E, OUTW, 0, 0, 0, 1.0f);
}
